// GatedAttention_74517682586450
// MI455X (gfx1250) — compile-verified, matrix operand order checked by reading, not run on hardware
//
#include <hip/hip_runtime.h>
#include <hip/hip_bf16.h>
#include <math.h>

// ---------------- problem constants ----------------
#define BATCH 4
#define SEQ   2048
#define HID   2048
#define NH    16
#define NKV   4
#define HD    128           // head dim D
#define GQA_G (NH / NKV)    // 4
#define SCALE 0.08838834764831845f   // 128^-0.5
#define EPS   1e-6f

typedef __bf16 bf16_t;
typedef __attribute__((ext_vector_type(16))) __bf16 v16bf;
typedef __attribute__((ext_vector_type(8)))  __bf16 v8bf;
typedef __attribute__((ext_vector_type(8)))  float  v8f;

// ---------------- helpers ----------------
// load 8 consecutive elements, convert to packed bf16x8
template<typename T>
__device__ __forceinline__ v8bf load8_cvt(const T* __restrict__ p)
{
    if constexpr (sizeof(T) == 2) {
        return *(const v8bf*)p;                     // 16B vector load
    } else {
        float4 a = *(const float4*)p;               // 2 x b128 loads
        float4 b = *(const float4*)(p + 4);
        v8bf r;
        r[0]=(bf16_t)a.x; r[1]=(bf16_t)a.y; r[2]=(bf16_t)a.z; r[3]=(bf16_t)a.w;
        r[4]=(bf16_t)b.x; r[5]=(bf16_t)b.y; r[6]=(bf16_t)b.z; r[7]=(bf16_t)b.w;
        return r;
    }
}

// fragment load: ISA 7.12.2 16-bit A/B layout. Lane (half = lane>>4, r = lane&15)
// holds row/col r with K = {half*8..half*8+7, half*8+16..half*8+23}:
// two contiguous 16B chunks at base and base+16 elements.
__device__ __forceinline__ v16bf frag_ld(const bf16_t* __restrict__ base)
{
    v8bf lo = *(const v8bf*)base;          // K: half*8 .. half*8+7
    v8bf hi = *(const v8bf*)(base + 16);   // K: half*8+16 .. half*8+23
    return __builtin_shufflevector(lo, hi, 0,1,2,3,4,5,6,7,8,9,10,11,12,13,14,15);
}

// ---------------- generic BF16-WMMA GEMM ----------------
// C[z] = alpha * A[z] (MxK) * B[z] (KxN, or NxK if TRANSB) + add[z]
// Block tile 64x128x32, 256 threads = 8 waves (2x4), wave tile 32x32 (2x2 WMMA).
#define BM 64
#define BN 128
#define BK 32
#define BKP 40   // padded LDS row (bf16 elems): 80B, 16B-aligned, bank-spread

template<typename TA, typename TB, bool TRANSB>
__global__ __launch_bounds__(256) void gemm_wmma(
    const TA* __restrict__ A, int lda, long long strideA,
    const TB* __restrict__ Bm, int ldb, long long strideB,
    float* __restrict__ C, int ldc, int cZdiv, long long strideC1, long long strideC2,
    const float* __restrict__ addM, int ldadd, long long strideAdd,
    int K, float alpha)
{
    __shared__ bf16_t As[BM * BKP];   // [m][k]   5120 B
    __shared__ bf16_t Bs[BN * BKP];   // [n][k]  10240 B

    const int tid  = threadIdx.x;
    const int lane = tid & 31;
    const int wave = tid >> 5;
    const int z    = blockIdx.z;

    const TA* Ab = A  + (long long)z * strideA;
    const TB* Bb = Bm + (long long)z * strideB;
    float*    Cb = C  + (long long)(z / cZdiv) * strideC1
                      + (long long)(z % cZdiv) * strideC2;
    const float* addB = addM ? (addM + (long long)z * strideAdd) : nullptr;

    const int m0 = blockIdx.y * BM;
    const int n0 = blockIdx.x * BN;
    const int waveM = (wave >> 2) * 32;   // 0 or 32
    const int waveN = (wave & 3)  * 32;   // 0,32,64,96

    // staging maps (all divisions exact for our shapes)
    const int a_mm = tid >> 2;            // 0..63
    const int a_kg = (tid & 3) * 8;       // 0,8,16,24
    const int bt_nn = tid >> 1;           // NT: 0..127
    const int bt_kg = (tid & 1) * 16;     // NT: 0 or 16
    const int bn_kk = tid >> 3;           // NN: 0..31
    const int bn_ng = (tid & 7) * 16;     // NN: 0..112

    v8f acc[2][2];
#pragma unroll
    for (int i = 0; i < 2; ++i)
#pragma unroll
        for (int j = 0; j < 2; ++j)
#pragma unroll
            for (int v = 0; v < 8; ++v) acc[i][j][v] = 0.0f;

    const int half = lane >> 4;
    const int r    = lane & 15;
    const bf16_t* aBase0 = &As[(waveM +  0 + r) * BKP + half * 8];
    const bf16_t* aBase1 = &As[(waveM + 16 + r) * BKP + half * 8];
    const bf16_t* bBase0 = &Bs[(waveN +  0 + r) * BKP + half * 8];
    const bf16_t* bBase1 = &Bs[(waveN + 16 + r) * BKP + half * 8];

    for (int k0 = 0; k0 < K; k0 += BK) {
        // ---- stage A tile (BM x BK): 8 consecutive k per thread, one b128 store
        {
            v8bf pk = load8_cvt(Ab + (long long)(m0 + a_mm) * lda + (k0 + a_kg));
            *(v8bf*)&As[a_mm * BKP + a_kg] = pk;
        }
        // ---- stage B tile as Bs[n][k]
        if (TRANSB) {
            // B row-major NxK: k is fast -> two vector loads + two b128 stores
            const TB* bp = Bb + (long long)(n0 + bt_nn) * ldb + (k0 + bt_kg);
            *(v8bf*)&Bs[bt_nn * BKP + bt_kg]     = load8_cvt(bp);
            *(v8bf*)&Bs[bt_nn * BKP + bt_kg + 8] = load8_cvt(bp + 8);
        } else {
            // B row-major KxN: n is fast -> vector global loads, transpose on LDS write
            const TB* bp = Bb + (long long)(k0 + bn_kk) * ldb + (n0 + bn_ng);
            v8bf g0 = load8_cvt(bp);
            v8bf g1 = load8_cvt(bp + 8);
#pragma unroll
            for (int j = 0; j < 8; ++j) {
                Bs[(bn_ng + j)     * BKP + bn_kk] = g0[j];
                Bs[(bn_ng + 8 + j) * BKP + bn_kk] = g1[j];
            }
        }
        __syncthreads();

        // ---- fragments: pure ds_load_b128 pairs, zero repack
        v16bf aF0 = frag_ld(aBase0);
        v16bf aF1 = frag_ld(aBase1);
        v16bf bF0 = frag_ld(bBase0);
        v16bf bF1 = frag_ld(bBase1);

        acc[0][0] = __builtin_amdgcn_wmma_f32_16x16x32_bf16(false, aF0, false, bF0, (short)0, acc[0][0], false, false);
        acc[0][1] = __builtin_amdgcn_wmma_f32_16x16x32_bf16(false, aF0, false, bF1, (short)0, acc[0][1], false, false);
        acc[1][0] = __builtin_amdgcn_wmma_f32_16x16x32_bf16(false, aF1, false, bF0, (short)0, acc[1][0], false, false);
        acc[1][1] = __builtin_amdgcn_wmma_f32_16x16x32_bf16(false, aF1, false, bF1, (short)0, acc[1][1], false, false);

        __syncthreads();

        // advance A/B along K
        aBase0 += 0; // LDS tiles are overwritten in place
    }

    // ---- epilogue: C layout per ISA (VGPR v -> row v + half*8; lane%16 -> col)
#pragma unroll
    for (int i = 0; i < 2; ++i) {
#pragma unroll
        for (int j = 0; j < 2; ++j) {
            const int colg = n0 + waveN + j * 16 + r;
#pragma unroll
            for (int v = 0; v < 8; ++v) {
                const int rowg = m0 + waveM + i * 16 + half * 8 + v;
                float val = acc[i][j][v] * alpha;
                if (addB) val += addB[(long long)rowg * ldadd + colg];
                Cb[(long long)rowg * ldc + colg] = val;
            }
        }
    }
}

// ---------------- fp32 -> bf16 convert ----------------
__global__ __launch_bounds__(256) void f32_to_bf16(const float* __restrict__ in,
                                                   bf16_t* __restrict__ out,
                                                   long long n)
{
    long long i = ((long long)blockIdx.x * 256 + threadIdx.x) * 8;
    if (i >= n) return;
    v8bf pk = load8_cvt(in + i);
    *(v8bf*)&out[i] = pk;
}

// ---------------- RMSNorm + RoPE (+GQA replicate) ----------------
// one 128-thread block per (b,s,h) row; dst layout [B, H*rep, S, D] bf16
__global__ __launch_bounds__(128) void qk_norm_rope(
    const float* __restrict__ src, long long rowStride, int headStride, int H,
    const float* __restrict__ normw,
    const float* __restrict__ cosT, const float* __restrict__ sinT,
    bf16_t* __restrict__ dst, int rep)
{
    long long bid = blockIdx.x;
    int h = (int)(bid % H);
    long long bs = bid / H;                 // b*SEQ + s
    int s = (int)(bs & (SEQ - 1));
    int b = (int)(bs >> 11);
    int d = threadIdx.x;

    const float* x = src + bs * rowStride + (long long)h * headStride;
    float v = x[d];

    __shared__ float red[128];
    red[d] = v * v;
    __syncthreads();
    for (int off = 64; off > 0; off >>= 1) {
        if (d < off) red[d] += red[d + off];
        __syncthreads();
    }
    float rms = rsqrtf(red[0] / (float)HD + EPS);

    float xn = v * rms * (1.0f + normw[d]);
    int pd = (d < 64) ? d + 64 : d - 64;
    float pn = x[pd] * rms * (1.0f + normw[pd]);
    float rot = (d < 64) ? -pn : pn;

    float c  = cosT[bs * HD + d];
    float sn = sinT[bs * HD + d];
    float outv = xn * c + rot * sn;

    int Htot = H * rep;
    for (int g = 0; g < rep; ++g) {
        long long ho = (long long)h * rep + g;
        dst[(((long long)b * Htot + ho) * SEQ + s) * HD + d] = (bf16_t)outv;
    }
}

// ---------------- V: [B,S,NKV,D] f32 -> [B,NH,S,D] bf16 (GQA replicate) ------
__global__ __launch_bounds__(256) void v_epilogue(const float* __restrict__ vproj,
                                                  bf16_t* __restrict__ v_bf)
{
    long long idx = (long long)blockIdx.x * 256 + threadIdx.x;   // B*S*NKV*D
    int d = (int)(idx & 127);
    int h = (int)((idx >> 7) & 3);
    int s = (int)((idx >> 9) & (SEQ - 1));
    int b = (int)(idx >> 20);
    float val = vproj[idx];
    for (int g = 0; g < GQA_G; ++g) {
        long long ho = h * GQA_G + g;
        v_bf[(((long long)b * NH + ho) * SEQ + s) * HD + d] = (bf16_t)val;
    }
}

// ---------------- row softmax over S=2048, in place ----------------
__global__ __launch_bounds__(256) void softmax_rows(float* __restrict__ attnw)
{
    long long row = blockIdx.x;                 // B*NH*S rows
    float* p = attnw + row * SEQ;
    int t = threadIdx.x;
    __shared__ float red[256];

    float vals[SEQ / 256];
    float mx = -3.4e38f;
#pragma unroll
    for (int i = 0; i < SEQ / 256; ++i) {
        vals[i] = p[i * 256 + t];
        mx = fmaxf(mx, vals[i]);
    }
    red[t] = mx; __syncthreads();
    for (int off = 128; off > 0; off >>= 1) {
        if (t < off) red[t] = fmaxf(red[t], red[t + off]);
        __syncthreads();
    }
    mx = red[0]; __syncthreads();

    float sum = 0.0f;
#pragma unroll
    for (int i = 0; i < SEQ / 256; ++i) {
        vals[i] = __expf(vals[i] - mx);
        sum += vals[i];
    }
    red[t] = sum; __syncthreads();
    for (int off = 128; off > 0; off >>= 1) {
        if (t < off) red[t] += red[t + off];
        __syncthreads();
    }
    float inv = 1.0f / red[0];
#pragma unroll
    for (int i = 0; i < SEQ / 256; ++i) p[i * 256 + t] = vals[i] * inv;
}

// ---------------- sigmoid gating: tmp[b,s,h*D+d] *= sigmoid(gate) ------------
__global__ __launch_bounds__(256) void gate_mul(float* __restrict__ tmp,
                                                const float* __restrict__ qg)
{
    long long idx = (long long)blockIdx.x * 256 + threadIdx.x;   // B*S*NH*D
    int d = (int)(idx & 127);
    int h = (int)((idx >> 7) & (NH - 1));
    long long bs = idx >> 11;                                    // b*SEQ + s
    float g = qg[(bs * NH + h) * (2 * HD) + HD + d];
    tmp[idx] *= 1.0f / (1.0f + __expf(-g));
}

// ---------------- host: pipeline launch ----------------
extern "C" void kernel_launch(void* const* d_in, const int* in_sizes, int n_in,
                              void* d_out, int out_size, void* d_ws, size_t ws_size,
                              hipStream_t stream)
{
    (void)in_sizes; (void)n_in; (void)out_size; (void)ws_size;

    const float* hs   = (const float*)d_in[0];
    const float* cosT = (const float*)d_in[1];
    const float* sinT = (const float*)d_in[2];
    const float* mask = (const float*)d_in[3];   // [S,S]
    const float* Wq   = (const float*)d_in[4];
    const float* Wk   = (const float*)d_in[5];
    const float* Wv   = (const float*)d_in[6];
    const float* Wo   = (const float*)d_in[7];
    const float* qnw  = (const float*)d_in[8];
    const float* knw  = (const float*)d_in[9];

    float* out_attn = (float*)d_out;                       // [B,S,NH*D]
    float* attnw    = out_attn + (long long)BATCH * SEQ * NH * HD; // [B,NH,S,S]

    // workspace carve-out
    char* w = (char*)d_ws;
    size_t off = 0;
    auto alloc = [&](size_t bytes) -> void* {
        void* p = w + off;
        off += (bytes + 255) & ~(size_t)255;
        return p;
    };
    const long long MS = (long long)BATCH * SEQ;   // 8192 token rows
    bf16_t* hs_bf = (bf16_t*)alloc(MS * HID * 2);
    bf16_t* Wq_bf = (bf16_t*)alloc((size_t)HID * (NH * 2 * HD) * 2);
    bf16_t* Wk_bf = (bf16_t*)alloc((size_t)HID * (NKV * HD) * 2);
    bf16_t* Wv_bf = (bf16_t*)alloc((size_t)HID * (NKV * HD) * 2);
    bf16_t* Wo_bf = (bf16_t*)alloc((size_t)(NH * HD) * HID * 2);
    float*  qg    = (float*)alloc(MS * (NH * 2 * HD) * 4);
    float*  kproj = (float*)alloc(MS * (NKV * HD) * 4);
    float*  vproj = (float*)alloc(MS * (NKV * HD) * 4);
    bf16_t* q_bf  = (bf16_t*)alloc((size_t)BATCH * NH * SEQ * HD * 2);
    bf16_t* k_bf  = (bf16_t*)alloc((size_t)BATCH * NH * SEQ * HD * 2);   // replicated
    bf16_t* v_bf  = (bf16_t*)alloc((size_t)BATCH * NH * SEQ * HD * 2);   // replicated
    float*  tmp   = (float*)alloc(MS * (NH * HD) * 4);

    const int BIG = 1 << 30;   // cZdiv for plain linear batched C

    // 1) bf16 conversions of reused operands (8 elems/thread, vectorized)
    {
        long long n;
        n = MS * HID;                 f32_to_bf16<<<(int)(n / 2048), 256, 0, stream>>>(hs, hs_bf, n);
        n = (long long)HID * NH*2*HD; f32_to_bf16<<<(int)(n / 2048), 256, 0, stream>>>(Wq, Wq_bf, n);
        n = (long long)HID * NKV*HD;  f32_to_bf16<<<(int)(n / 2048), 256, 0, stream>>>(Wk, Wk_bf, n);
        n = (long long)HID * NKV*HD;  f32_to_bf16<<<(int)(n / 2048), 256, 0, stream>>>(Wv, Wv_bf, n);
        n = (long long)(NH*HD) * HID; f32_to_bf16<<<(int)(n / 2048), 256, 0, stream>>>(Wo, Wo_bf, n);
    }

    // 2) projections: qg = hs@Wq (8192x4096), kproj = hs@Wk, vproj = hs@Wv
    gemm_wmma<bf16_t, bf16_t, false><<<dim3((NH*2*HD)/BN, MS/BM, 1), 256, 0, stream>>>(
        hs_bf, HID, 0, Wq_bf, NH*2*HD, 0,
        qg, NH*2*HD, BIG, 0, 0, nullptr, 0, 0, HID, 1.0f);
    gemm_wmma<bf16_t, bf16_t, false><<<dim3((NKV*HD)/BN, MS/BM, 1), 256, 0, stream>>>(
        hs_bf, HID, 0, Wk_bf, NKV*HD, 0,
        kproj, NKV*HD, BIG, 0, 0, nullptr, 0, 0, HID, 1.0f);
    gemm_wmma<bf16_t, bf16_t, false><<<dim3((NKV*HD)/BN, MS/BM, 1), 256, 0, stream>>>(
        hs_bf, HID, 0, Wv_bf, NKV*HD, 0,
        vproj, NKV*HD, BIG, 0, 0, nullptr, 0, 0, HID, 1.0f);

    // 3) RMSNorm + RoPE epilogues (k/v replicated across GQA groups)
    qk_norm_rope<<<(int)(MS * NH), 128, 0, stream>>>(
        qg, NH*2*HD, 2*HD, NH, qnw, cosT, sinT, q_bf, 1);
    qk_norm_rope<<<(int)(MS * NKV), 128, 0, stream>>>(
        kproj, NKV*HD, HD, NKV, knw, cosT, sinT, k_bf, GQA_G);
    v_epilogue<<<(int)((MS * NKV * HD) / 256), 256, 0, stream>>>(vproj, v_bf);

    // 4) scores = SCALE * Q K^T + mask  -> attnw  (batched over B*NH)
    gemm_wmma<bf16_t, bf16_t, true><<<dim3(SEQ/BN, SEQ/BM, BATCH*NH), 256, 0, stream>>>(
        q_bf, HD, (long long)SEQ * HD,
        k_bf, HD, (long long)SEQ * HD,
        attnw, SEQ, BIG, 0, (long long)SEQ * SEQ,
        mask, SEQ, 0, HD, SCALE);

    // 5) softmax rows (in place in d_out attn_weights region)
    softmax_rows<<<(int)((long long)BATCH * NH * SEQ), 256, 0, stream>>>(attnw);

    // 6) PV: tmp[b, s, h*D+d] = P @ V   (fp32 P converted to bf16 on the fly)
    gemm_wmma<float, bf16_t, false><<<dim3(HD/BN ? HD/BN : 1, SEQ/BM, BATCH*NH), 256, 0, stream>>>(
        attnw, SEQ, (long long)SEQ * SEQ,
        v_bf, HD, (long long)SEQ * HD,
        tmp, NH*HD, NH, (long long)SEQ * NH * HD, HD,
        nullptr, 0, 0, SEQ, 1.0f);

    // 7) sigmoid gating
    gate_mul<<<(int)((MS * NH * HD) / 256), 256, 0, stream>>>(tmp, qg);

    // 8) attn_output = gated @ Wo
    gemm_wmma<float, bf16_t, false><<<dim3(HID/BN, MS/BM, 1), 256, 0, stream>>>(
        tmp, NH*HD, 0, Wo_bf, HID, 0,
        out_attn, HID, BIG, 0, 0, nullptr, 0, 0, NH*HD, 1.0f);
}
